// GraphAttentionLayer_P0_8632884265509
// MI455X (gfx1250) — matrix operand order repaired and checked by reading, not compiled, not run on hardware
//
#include <hip/hip_runtime.h>

#define N_NODES 8192
#define IN_CH   256
#define OUT_CH  64

typedef __attribute__((ext_vector_type(16))) _Float16 v16h;
typedef __attribute__((ext_vector_type(8)))  float    v8f;
typedef __attribute__((ext_vector_type(4)))  int      i32x4;
typedef __attribute__((ext_vector_type(4)))  float    f32x4;

// ---------------------------------------------------------------------------
// Kernel 1: Wh = h @ W (fp32 accumulate). Store WhT as f16 [OUT_CH][N] for
// WMMA B-tiles, and per-row attention projections Wh1 = Wh@a[:64], Wh2 = Wh@a[64:].
// One block (64 threads) per node row; h row staged in LDS.
// ---------------------------------------------------------------------------
__global__ void __launch_bounds__(64)
k_proj(const float* __restrict__ h, const float* __restrict__ W,
       const float* __restrict__ a, _Float16* __restrict__ WhT,
       float* __restrict__ Wh1, float* __restrict__ Wh2) {
    __shared__ float sh[IN_CH];
    __shared__ float r1[64], r2[64];
    const int row = blockIdx.x;
    const int t   = threadIdx.x;
    for (int k = t; k < IN_CH; k += 64) sh[k] = h[(size_t)row * IN_CH + k];
    __syncthreads();
    float acc = 0.f;
#pragma unroll 8
    for (int k = 0; k < IN_CH; ++k) acc = fmaf(sh[k], W[k * OUT_CH + t], acc);
    WhT[(size_t)t * N_NODES + row] = (_Float16)acc;
    r1[t] = acc * a[t];
    r2[t] = acc * a[OUT_CH + t];
    __syncthreads();
    for (int s = 32; s > 0; s >>= 1) {
        if (t < s) { r1[t] += r1[t + s]; r2[t] += r2[t + s]; }
        __syncthreads();
    }
    if (t == 0) { Wh1[row] = r1[0]; Wh2[row] = r2[0]; }
}

// ---------------------------------------------------------------------------
// Kernel 2: global max of Wh2 (upper bound for single-pass softmax max).
// ---------------------------------------------------------------------------
__global__ void __launch_bounds__(256)
k_gmax(const float* __restrict__ Wh2, float* __restrict__ gmax) {
    __shared__ float s[256];
    float m = -3.0e38f;
    for (int j = (int)threadIdx.x; j < N_NODES; j += 256) m = fmaxf(m, Wh2[j]);
    s[threadIdx.x] = m;
    __syncthreads();
    for (int st = 128; st > 0; st >>= 1) {
        if ((int)threadIdx.x < st) s[threadIdx.x] = fmaxf(s[threadIdx.x], s[threadIdx.x + st]);
        __syncthreads();
    }
    if (threadIdx.x == 0) *gmax = s[0];
}

// ---------------------------------------------------------------------------
// Kernel 3: fused masked-softmax attention  out = relu( softmax(mask(e)) @ Wh ).
// Block = 256 threads (8 waves) handles 16 output rows. Each wave streams
// j-chunks of 32 (stride 8*32), builds the 16x32 f16 probability tile directly
// in WMMA A-layout, and accumulates P @ Wh with v_wmma_f32_16x16x32_f16 into
// four 16x16 f32 accumulators (N = 64). adj is streamed nontemporally (read
// once, 256 MB dominant traffic). Cross-wave (acc, l) reduction via LDS.
// ---------------------------------------------------------------------------
__global__ void __launch_bounds__(256)
k_gat(const int* __restrict__ adj, const _Float16* __restrict__ WhT,
      const float* __restrict__ Wh1, const float* __restrict__ Wh2,
      const float* __restrict__ gmaxp, float* __restrict__ out) {
    __shared__ float s_acc[8][16][64];   // per-wave partial h_prime tiles
    __shared__ float s_l[8][16];         // per-wave partial softmax denominators

    const int rb     = blockIdx.x;
    const int lane   = threadIdx.x & 31;
    const int wave   = threadIdx.x >> 5;
    const int ilocal = lane & 15;        // A-matrix row / B-matrix column in tile
    const int hi     = lane >> 4;        // lane half selects K sub-block
    const int row    = rb * 16 + ilocal;

    const float wh1_i = Wh1[row];
    const float gmax  = *gmaxp;
    float Mi = wh1_i + gmax;             // lrelu is monotone -> this bounds row max
    Mi = Mi >= 0.f ? Mi : 0.5f * Mi;

    v8f acc0 = {}, acc1 = {}, acc2 = {}, acc3 = {};
    float lsum = 0.f;

    const int  kb      = hi * 8;         // A layout: K runs [kb..kb+7],[kb+16..kb+23]
    const int* adj_row = adj + (size_t)row * N_NODES;

    for (int jc = wave * 32; jc < N_NODES; jc += 8 * 32) {
        const int j0 = jc + kb;
        const int j1 = j0 + 16;

        // Stream adjacency (read-once): nontemporal 128-bit loads.
        i32x4 a0 = __builtin_nontemporal_load((const i32x4*)(adj_row + j0));
        i32x4 a1 = __builtin_nontemporal_load((const i32x4*)(adj_row + j0 + 4));
        i32x4 a2 = __builtin_nontemporal_load((const i32x4*)(adj_row + j1));
        i32x4 a3 = __builtin_nontemporal_load((const i32x4*)(adj_row + j1 + 4));

        f32x4 w0 = *(const f32x4*)(Wh2 + j0);
        f32x4 w1 = *(const f32x4*)(Wh2 + j0 + 4);
        f32x4 w2 = *(const f32x4*)(Wh2 + j1);
        f32x4 w3 = *(const f32x4*)(Wh2 + j1 + 4);

        int   av[16];
        float wv[16];
#pragma unroll
        for (int u = 0; u < 4; ++u) {
            av[u]      = a0[u];  wv[u]      = w0[u];
            av[u + 4]  = a1[u];  wv[u + 4]  = w1[u];
            av[u + 8]  = a2[u];  wv[u + 8]  = w2[u];
            av[u + 12] = a3[u];  wv[u + 12] = w3[u];
        }

        // Masked LeakyReLU scores -> unnormalized probs, packed in A layout:
        // element u -> K = kb + u (u<8), K = kb + 16 + (u-8) (u>=8).
        union { v16h v; _Float16 e[16]; } A;
#pragma unroll
        for (int u = 0; u < 16; ++u) {
            float e = wh1_i + wv[u];
            e = e >= 0.f ? e : 0.5f * e;
            float p = __expf(e - Mi);
            p = av[u] > 0 ? p : 0.f;
            lsum += p;
            A.e[u] = (_Float16)p;
        }

        // B tiles: 32(K) x 16(N) f16; lane = column n, lane-half h = K [8h..8h+7],[16+8h..16+8h+7].
        // WhT is [OUT_CH][N] so each lane loads two runs of 8 contiguous halves, 16 apart.
#pragma unroll
        for (int nt = 0; nt < 4; ++nt) {
            union { v16h v; f32x4 f4[2]; } B;
            const f32x4* bp =
                (const f32x4*)(WhT + (size_t)(nt * 16 + ilocal) * N_NODES + jc + hi * 8);
            B.f4[0] = bp[0];
            B.f4[1] = bp[2];
            if      (nt == 0) acc0 = __builtin_amdgcn_wmma_f32_16x16x32_f16(false, A.v, false, B.v, (short)0, acc0, false, false);
            else if (nt == 1) acc1 = __builtin_amdgcn_wmma_f32_16x16x32_f16(false, A.v, false, B.v, (short)0, acc1, false, false);
            else if (nt == 2) acc2 = __builtin_amdgcn_wmma_f32_16x16x32_f16(false, A.v, false, B.v, (short)0, acc2, false, false);
            else              acc3 = __builtin_amdgcn_wmma_f32_16x16x32_f16(false, A.v, false, B.v, (short)0, acc3, false, false);
        }
    }

    // C/D layout: lane L, VGPR r -> row (r + 8*hi), col (nt*16 + ilocal).
#pragma unroll
    for (int r = 0; r < 8; ++r) {
        s_acc[wave][r + 8 * hi][ 0 + ilocal] = acc0[r];
        s_acc[wave][r + 8 * hi][16 + ilocal] = acc1[r];
        s_acc[wave][r + 8 * hi][32 + ilocal] = acc2[r];
        s_acc[wave][r + 8 * hi][48 + ilocal] = acc3[r];
    }
    lsum += __shfl_xor(lsum, 16, 32);   // combine the two lane-halves of each row
    if (hi == 0) s_l[wave][ilocal] = lsum;
    __syncthreads();

    // Final cross-wave reduce, normalize, ReLU, store.
    for (int pos = (int)threadIdx.x; pos < 16 * 64; pos += 256) {
        const int r = pos >> 6;
        const int c = pos & 63;
        float sum = 0.f, lt = 0.f;
#pragma unroll
        for (int w = 0; w < 8; ++w) { sum += s_acc[w][r][c]; lt += s_l[w][r]; }
        float o = sum / fmaxf(lt, 1e-30f);
        out[(size_t)(rb * 16 + r) * OUT_CH + c] = fmaxf(o, 0.f);
    }
}

// ---------------------------------------------------------------------------
extern "C" void kernel_launch(void* const* d_in, const int* in_sizes, int n_in,
                              void* d_out, int out_size, void* d_ws, size_t ws_size,
                              hipStream_t stream) {
    const float* h   = (const float*)d_in[0];
    const int*   adj = (const int*)  d_in[1];
    const float* W   = (const float*)d_in[2];
    const float* a   = (const float*)d_in[3];
    float*       out = (float*)d_out;

    // Workspace layout: WhT f16 [64][8192] | Wh1 f32[8192] | Wh2 f32[8192] | gmax f32
    _Float16* WhT  = (_Float16*)d_ws;
    float*    Wh1  = (float*)((char*)d_ws + sizeof(_Float16) * OUT_CH * N_NODES);
    float*    Wh2  = Wh1 + N_NODES;
    float*    gmax = Wh2 + N_NODES;

    k_proj<<<N_NODES, 64, 0, stream>>>(h, W, a, WhT, Wh1, Wh2);
    k_gmax<<<1, 256, 0, stream>>>(Wh2, gmax);
    k_gat<<<N_NODES / 16, 256, 0, stream>>>(adj, WhT, Wh1, Wh2, gmax, out);
}
